// Hyper_Block_54125177864653
// MI455X (gfx1250) — compile-verified, matrix operand order checked by reading, not run on hardware
//
#include <hip/hip_runtime.h>
#include <hip/hip_bf16.h>

typedef __bf16 bf16_t;
typedef __attribute__((ext_vector_type(16))) __bf16 v16bf;
typedef __attribute__((ext_vector_type(8)))  __bf16 v8bf;
typedef __attribute__((ext_vector_type(8)))  float  v8f;

__device__ inline v8f zero8() {
  v8f z = {0.f, 0.f, 0.f, 0.f, 0.f, 0.f, 0.f, 0.f};
  return z;
}

// A/B fragment loader for v_wmma_f32_16x16x32_bf16.
// Source is row-major-in-K: lane holds row/col (lane&15); per ISA layout the
// lane's 16 elements are two contiguous K-runs of 8: [8g .. 8g+7] and
// [16+8g .. 23+8g] where g = lane>>4.  rowptr points at k0 of that row.
// Lowers to two global_load_b128 per fragment.
__device__ inline v16bf load_frag(const bf16_t* rowptr, int g) {
  v8bf lo = *(const v8bf*)(rowptr + 8 * g);
  v8bf hi = *(const v8bf*)(rowptr + 16 + 8 * g);
  v16bf f;
#pragma unroll
  for (int i = 0; i < 8; ++i) { f[i] = lo[i]; f[8 + i] = hi[i]; }
  return f;
}

__device__ inline v8f wmma_bf16(v16bf a, v16bf b, v8f c) {
  return __builtin_amdgcn_wmma_f32_16x16x32_bf16(false, a, false, b, (short)0, c,
                                                 false, false);
}

// 64x64 wave-tile GEMM core: acc[i][j] += A(64xK) @ B(64xK)^T tile.
// Abase points at (row64+nn) row 0 offset; rows strided lda. Bbase likewise.
// 16 WMMAs per 16 b128 loads per k-step -> 2x the compute density of 16x64.
__device__ inline void gemm_tile_64x64(const bf16_t* __restrict__ Abase, int lda,
                                       const bf16_t* __restrict__ Bbase, int ldb,
                                       int K, int g, v8f acc[4][4]) {
#pragma unroll
  for (int i = 0; i < 4; ++i)
#pragma unroll
    for (int j = 0; j < 4; ++j) acc[i][j] = zero8();
  for (int k0 = 0; k0 < K; k0 += 32) {
    v16bf a[4];
#pragma unroll
    for (int i = 0; i < 4; ++i)
      a[i] = load_frag(Abase + (size_t)(i * 16) * lda + k0, g);
#pragma unroll
    for (int j = 0; j < 4; ++j) {
      v16bf b = load_frag(Bbase + (size_t)(j * 16) * ldb + k0, g);
#pragma unroll
      for (int i = 0; i < 4; ++i) acc[i][j] = wmma_bf16(a[i], b, acc[i][j]);
    }
  }
}

// ---------------------------------------------------------------- casts
__global__ void cast_f32_bf16(const float* __restrict__ in, bf16_t* __restrict__ out, int n) {
  int i = blockIdx.x * 256 + threadIdx.x;
  if (i < n) out[i] = (bf16_t)in[i];
}

__global__ void transpose_cast_M(const float* __restrict__ M, bf16_t* __restrict__ MT) {
  int i = blockIdx.x * 256 + threadIdx.x;  // 192*192
  if (i < 192 * 192) {
    int d = i / 192, e = i % 192;
    MT[d * 192 + e] = (bf16_t)M[e * 192 + d];
  }
}

// ---------------------------------------------------------------- LN1 + cast
__global__ __launch_bounds__(256) void ln1_cast(const float* __restrict__ x,
                                                const float* __restrict__ w,
                                                const float* __restrict__ b,
                                                bf16_t* __restrict__ h) {
  int row = blockIdx.x, t = threadIdx.x;
  const float* xr = x + (size_t)row * 768;
  float v[3], s = 0.f, s2 = 0.f;
#pragma unroll
  for (int i = 0; i < 3; ++i) { v[i] = xr[t + 256 * i]; s += v[i]; s2 += v[i] * v[i]; }
#pragma unroll
  for (int m = 16; m >= 1; m >>= 1) { s += __shfl_xor(s, m, 32); s2 += __shfl_xor(s2, m, 32); }
  __shared__ float red[16];
  int wv = t >> 5, ln = t & 31;
  if (ln == 0) { red[wv] = s; red[8 + wv] = s2; }
  __syncthreads();
  float ts = 0.f, ts2 = 0.f;
#pragma unroll
  for (int i = 0; i < 8; ++i) { ts += red[i]; ts2 += red[8 + i]; }
  float mu = ts * (1.f / 768.f);
  float var = ts2 * (1.f / 768.f) - mu * mu;
  float rstd = rsqrtf(var + 1e-5f);
  bf16_t* hr = h + (size_t)row * 768;
#pragma unroll
  for (int i = 0; i < 3; ++i) {
    int c = t + 256 * i;
    hr[c] = (bf16_t)((v[i] - mu) * rstd * w[c] + b[c]);
  }
}

// ---------------------------------------------------------------- QKV GEMM
// out[8192,2304](bf16) = A[8192,768] @ W[2304,768]^T ; wave tile 64x64
__global__ __launch_bounds__(256) void gemm_qkv(const bf16_t* __restrict__ A,
                                                const bf16_t* __restrict__ W,
                                                bf16_t* __restrict__ out) {
  int wave = (blockIdx.x * 256 + threadIdx.x) >> 5;
  int lane = threadIdx.x & 31;
  const int NT = 2304 / 64;  // 36
  int it = wave / NT, jt = wave % NT;  // it < 128
  int row64 = it * 64, col64 = jt * 64;
  int g = lane >> 4, nn = lane & 15;
  v8f acc[4][4];
  gemm_tile_64x64(A + (size_t)(row64 + nn) * 768, 768,
                  W + (size_t)(col64 + nn) * 768, 768, 768, g, acc);
#pragma unroll
  for (int i = 0; i < 4; ++i)
#pragma unroll
    for (int j = 0; j < 4; ++j)
#pragma unroll
      for (int r = 0; r < 8; ++r) {
        int row = row64 + i * 16 + r + 8 * g, col = col64 + j * 16 + nn;
        out[(size_t)row * 2304 + col] = (bf16_t)acc[i][j][r];
      }
}

// ---------------------------------------------------------------- qM / kM GEMM
// per (b,h): [1024x192] (q or k slice of qkv) @ M -> out bf16 (B = MT rows)
__global__ __launch_bounds__(256) void qm_gemm(const bf16_t* __restrict__ qkv,
                                               const bf16_t* __restrict__ MT,
                                               bf16_t* __restrict__ outQ, int sec) {
  int wave = (blockIdx.x * 256 + threadIdx.x) >> 5;
  int lane = threadIdx.x & 31;
  const int PER_BH = 16 * 3;  // 16 row64-tiles x 3 col64 groups
  int bh = wave / PER_BH, rem = wave % PER_BH;
  int it = rem / 3, jt = rem % 3;
  int b = bh >> 2, hh = bh & 3;
  int row64 = it * 64, col64 = jt * 64;
  int g = lane >> 4, nn = lane & 15;
  v8f acc[4][4];
  gemm_tile_64x64(qkv + (size_t)(b * 1024 + row64 + nn) * 2304 + sec * 768 + hh * 192,
                  2304, MT + (size_t)(col64 + nn) * 192, 192, 192, g, acc);
#pragma unroll
  for (int i = 0; i < 4; ++i)
#pragma unroll
    for (int j = 0; j < 4; ++j)
#pragma unroll
      for (int r = 0; r < 8; ++r) {
        int row = row64 + i * 16 + r + 8 * g, col = col64 + j * 16 + nn;
        outQ[((size_t)bh * 1024 + row) * 192 + col] = (bf16_t)acc[i][j][r];
      }
}

// ---------------------------------------------------------------- kMk = k . (kM)
__global__ __launch_bounds__(256) void kmk_dot(const bf16_t* __restrict__ qkv,
                                               const bf16_t* __restrict__ kM,
                                               float* __restrict__ kMk) {
  int wave = (blockIdx.x * 256 + threadIdx.x) >> 5;  // 32768 rows
  int lane = threadIdx.x & 31;
  int bh = wave >> 10, n = wave & 1023;
  int b = bh >> 2, hh = bh & 3;
  const bf16_t* kr = qkv + (size_t)(b * 1024 + n) * 2304 + 768 + hh * 192;
  const bf16_t* mr = kM + ((size_t)bh * 1024 + n) * 192;
  float s = 0.f;
  for (int i = lane; i < 192; i += 32) s += (float)kr[i] * (float)mr[i];
#pragma unroll
  for (int m = 16; m >= 1; m >>= 1) s += __shfl_xor(s, m, 32);
  if (lane == 0) kMk[bh * 1024 + n] = s;
}

// ---------------------------------------------------------------- V transpose
__global__ __launch_bounds__(256) void v_transpose(const bf16_t* __restrict__ qkv,
                                                   bf16_t* __restrict__ vT) {
  int blk = blockIdx.x;  // 32 * 64 * 12
  int dt = blk % 12, nk = (blk / 12) % 64, bh = blk / (12 * 64);
  int b = bh >> 2, hh = bh & 3;
  int t = threadIdx.x, tx = t & 15, ty = t >> 4;
  __shared__ bf16_t tile[16][17];
  tile[ty][tx] =
      qkv[(size_t)(b * 1024 + nk * 16 + ty) * 2304 + 1536 + hh * 192 + dt * 16 + tx];
  __syncthreads();
  vT[((size_t)bh * 192 + dt * 16 + ty) * 1024 + nk * 16 + tx] = tile[tx][ty];
}

// ---------------------------------------------------------------- flash attention
// wave = one 16-row q tile of one (b,h); online softmax over 32-key blocks.
// logits = 2*scale*(qM.k) - scale*kMk  (qMq cancels inside softmax).
__global__ __launch_bounds__(256) void flash_attn(const bf16_t* __restrict__ qkv,
                                                  const bf16_t* __restrict__ qM,
                                                  const float* __restrict__ kMk,
                                                  const bf16_t* __restrict__ vT,
                                                  const float* __restrict__ scale_p,
                                                  bf16_t* __restrict__ ybuf) {
  int wv = threadIdx.x >> 5, lane = threadIdx.x & 31;
  int bh = blockIdx.x >> 3;
  int qt = ((blockIdx.x & 7) << 3) + wv;  // 0..63
  int b = bh >> 2, hh = bh & 3;
  int g = lane >> 4, nn = lane & 15;
  float scale = scale_p[0];
  float twos = 2.f * scale;

  v16bf aM[6];
  const bf16_t* qMrow = qM + ((size_t)bh * 1024 + qt * 16 + nn) * 192;
#pragma unroll
  for (int t = 0; t < 6; ++t) aM[t] = load_frag(qMrow + 32 * t, g);

  v8f acc[12];
#pragma unroll
  for (int t = 0; t < 12; ++t) acc[t] = zero8();
  float mrow[8], lrow[8];
#pragma unroll
  for (int r = 0; r < 8; ++r) { mrow[r] = -1e30f; lrow[r] = 0.f; }

  __shared__ bf16_t pbuf[8][16][32];  // per-wave P tile (C-layout -> A-layout)
  const bf16_t* kbase = qkv + (size_t)b * 1024 * 2304 + 768 + hh * 192;
  const bf16_t* vTbase = vT + (size_t)bh * 192 * 1024;
  const float* kmkB = kMk + bh * 1024;

  for (int kb = 0; kb < 1024; kb += 32) {
    v8f s0 = zero8(), s1 = zero8();
    const bf16_t* k0p = kbase + (size_t)(kb + nn) * 2304;
    const bf16_t* k1p = kbase + (size_t)(kb + 16 + nn) * 2304;
#pragma unroll
    for (int t = 0; t < 6; ++t) {
      v16bf b0 = load_frag(k0p + 32 * t, g);
      s0 = wmma_bf16(aM[t], b0, s0);
      v16bf b1 = load_frag(k1p + 32 * t, g);
      s1 = wmma_bf16(aM[t], b1, s1);
    }
    float km0 = kmkB[kb + nn] * scale;
    float km1 = kmkB[kb + 16 + nn] * scale;
    float p0[8], p1[8], alpha[8];
#pragma unroll
    for (int r = 0; r < 8; ++r) {
      float a0 = twos * s0[r] - km0;
      float a1 = twos * s1[r] - km1;
      float tm = fmaxf(a0, a1);
      tm = fmaxf(tm, __shfl_xor(tm, 1, 32));
      tm = fmaxf(tm, __shfl_xor(tm, 2, 32));
      tm = fmaxf(tm, __shfl_xor(tm, 4, 32));
      tm = fmaxf(tm, __shfl_xor(tm, 8, 32));
      float mn = fmaxf(mrow[r], tm);
      alpha[r] = __expf(mrow[r] - mn);
      p0[r] = __expf(a0 - mn);
      p1[r] = __expf(a1 - mn);
      float rs = p0[r] + p1[r];
      rs += __shfl_xor(rs, 1, 32);
      rs += __shfl_xor(rs, 2, 32);
      rs += __shfl_xor(rs, 4, 32);
      rs += __shfl_xor(rs, 8, 32);
      lrow[r] = lrow[r] * alpha[r] + rs;
      mrow[r] = mn;
    }
#pragma unroll
    for (int t = 0; t < 12; ++t)
#pragma unroll
      for (int r = 0; r < 8; ++r) acc[t][r] *= alpha[r];
    // scatter P (C layout) into LDS, gather back in A layout
#pragma unroll
    for (int r = 0; r < 8; ++r) {
      pbuf[wv][r + 8 * g][nn] = (bf16_t)p0[r];
      pbuf[wv][r + 8 * g][16 + nn] = (bf16_t)p1[r];
    }
    asm volatile("s_wait_dscnt 0" ::: "memory");  // wave-synchronous LDS turnaround
    v16bf pA;
    {
      const bf16_t* pr = &pbuf[wv][nn][0];
      v8bf lo = *(const v8bf*)(pr + 8 * g);
      v8bf hi = *(const v8bf*)(pr + 16 + 8 * g);
#pragma unroll
      for (int i = 0; i < 8; ++i) { pA[i] = lo[i]; pA[8 + i] = hi[i]; }
    }
#pragma unroll
    for (int t = 0; t < 12; ++t) {
      v16bf bv = load_frag(vTbase + (size_t)(t * 16 + nn) * 1024 + kb, g);
      acc[t] = wmma_bf16(pA, bv, acc[t]);
    }
  }
#pragma unroll
  for (int r = 0; r < 8; ++r) {
    float inv = 1.f / lrow[r];
    int row = qt * 16 + r + 8 * g;
    bf16_t* yr = ybuf + ((size_t)(b * 1024) + row) * 768 + hh * 192;
#pragma unroll
    for (int t = 0; t < 12; ++t) yr[t * 16 + nn] = (bf16_t)(acc[t][r] * inv);
  }
}

// ---------------------------------------------------------------- proj + residual
__global__ __launch_bounds__(256) void gemm_proj(const bf16_t* __restrict__ Y,
                                                 const bf16_t* __restrict__ Wp,
                                                 const float* __restrict__ pb,
                                                 const float* __restrict__ x,
                                                 float* __restrict__ out) {
  int wave = (blockIdx.x * 256 + threadIdx.x) >> 5;
  int lane = threadIdx.x & 31;
  const int NT = 768 / 64;  // 12
  int it = wave / NT, jt = wave % NT;  // it < 128
  int row64 = it * 64, col64 = jt * 64;
  int g = lane >> 4, nn = lane & 15;
  v8f acc[4][4];
  gemm_tile_64x64(Y + (size_t)(row64 + nn) * 768, 768,
                  Wp + (size_t)(col64 + nn) * 768, 768, 768, g, acc);
#pragma unroll
  for (int i = 0; i < 4; ++i)
#pragma unroll
    for (int j = 0; j < 4; ++j)
#pragma unroll
      for (int r = 0; r < 8; ++r) {
        int row = row64 + i * 16 + r + 8 * g, col = col64 + j * 16 + nn;
        out[(size_t)row * 768 + col] =
            acc[i][j][r] + pb[col] + x[(size_t)row * 768 + col];
      }
}

// ---------------------------------------------------------------- LN2 + RBF MLP
__global__ __launch_bounds__(256) void ln2_rbf(float* __restrict__ out,
                                               const float* __restrict__ w,
                                               const float* __restrict__ bb,
                                               const float* __restrict__ centers,
                                               const float* __restrict__ beta,
                                               const float* __restrict__ fc_w) {
  int row = blockIdx.x, t = threadIdx.x;
  __shared__ float h2[768];
  __shared__ float red[16];
  __shared__ float rbf[20];
  const float* xr = out + (size_t)row * 768;
  float v[3], s = 0.f, s2 = 0.f;
#pragma unroll
  for (int i = 0; i < 3; ++i) { v[i] = xr[t + 256 * i]; s += v[i]; s2 += v[i] * v[i]; }
#pragma unroll
  for (int m = 16; m >= 1; m >>= 1) { s += __shfl_xor(s, m, 32); s2 += __shfl_xor(s2, m, 32); }
  int wv = t >> 5, ln = t & 31;
  if (ln == 0) { red[wv] = s; red[8 + wv] = s2; }
  __syncthreads();
  float ts = 0.f, ts2 = 0.f;
#pragma unroll
  for (int i = 0; i < 8; ++i) { ts += red[i]; ts2 += red[8 + i]; }
  float mu = ts * (1.f / 768.f);
  float var = ts2 * (1.f / 768.f) - mu * mu;
  float rstd = rsqrtf(var + 1e-5f);
  float hv[3], pa = 0.f;
#pragma unroll
  for (int i = 0; i < 3; ++i) {
    int c = t + 256 * i;
    hv[i] = (v[i] - mu) * rstd * w[c] + bb[c];
    h2[c] = hv[i];
    pa += hv[i] * hv[i];
  }
#pragma unroll
  for (int m = 16; m >= 1; m >>= 1) pa += __shfl_xor(pa, m, 32);
  __syncthreads();
  if (ln == 0) red[wv] = pa;
  __syncthreads();
  float a2 = 0.f;
#pragma unroll
  for (int i = 0; i < 8; ++i) a2 += red[i];
  for (int j = wv; j < 20; j += 8) {
    const float* cj = centers + (size_t)j * 768;
    float dot = 0.f, cn = 0.f;
    for (int i = ln; i < 768; i += 32) { float cv = cj[i]; dot += h2[i] * cv; cn += cv * cv; }
#pragma unroll
    for (int m = 16; m >= 1; m >>= 1) { dot += __shfl_xor(dot, m, 32); cn += __shfl_xor(cn, m, 32); }
    if (ln == 0) rbf[j] = __expf(-beta[j] * (a2 - 2.f * dot + cn));
  }
  __syncthreads();
#pragma unroll
  for (int i = 0; i < 3; ++i) {
    int c = t + 256 * i;
    float a = 0.f;
#pragma unroll
    for (int j = 0; j < 20; ++j) a += rbf[j] * fc_w[c * 20 + j];
    out[(size_t)row * 768 + c] = v[i] + a;
  }
}

// ---------------------------------------------------------------- launch
extern "C" void kernel_launch(void* const* d_in, const int* in_sizes, int n_in,
                              void* d_out, int out_size, void* d_ws, size_t ws_size,
                              hipStream_t stream) {
  const float* x = (const float*)d_in[0];
  const float* n1w = (const float*)d_in[1];
  const float* n1b = (const float*)d_in[2];
  const float* qkv_w = (const float*)d_in[3];
  const float* M = (const float*)d_in[4];
  const float* scale = (const float*)d_in[5];
  const float* proj_w = (const float*)d_in[6];
  const float* proj_b = (const float*)d_in[7];
  const float* n2w = (const float*)d_in[8];
  const float* n2b = (const float*)d_in[9];
  const float* centers = (const float*)d_in[10];
  const float* beta = (const float*)d_in[11];
  const float* fc_w = (const float*)d_in[12];
  float* out = (float*)d_out;

  char* ws = (char*)d_ws;
  size_t off = 0;
  auto take = [&](size_t bytes) -> char* {
    char* p = ws + off;
    off += (bytes + 255) & ~(size_t)255;
    return p;
  };
  bf16_t* hbuf  = (bf16_t*)take(8192ull * 768 * 2);
  bf16_t* wqkvb = (bf16_t*)take(2304ull * 768 * 2);
  bf16_t* wprjb = (bf16_t*)take(768ull * 768 * 2);
  bf16_t* MTb   = (bf16_t*)take(192ull * 192 * 2);
  bf16_t* qkvb  = (bf16_t*)take(8192ull * 2304 * 2);
  bf16_t* qMb   = (bf16_t*)take(32ull * 1024 * 192 * 2);
  bf16_t* kMb   = (bf16_t*)take(32ull * 1024 * 192 * 2);
  float*  kMkb  = (float*)take(32ull * 1024 * 4);
  bf16_t* vTb   = (bf16_t*)take(32ull * 192 * 1024 * 2);
  bf16_t* ybuf  = (bf16_t*)take(8192ull * 768 * 2);

  cast_f32_bf16<<<(2304 * 768 + 255) / 256, 256, 0, stream>>>(qkv_w, wqkvb, 2304 * 768);
  cast_f32_bf16<<<(768 * 768 + 255) / 256, 256, 0, stream>>>(proj_w, wprjb, 768 * 768);
  transpose_cast_M<<<(192 * 192 + 255) / 256, 256, 0, stream>>>(M, MTb);
  ln1_cast<<<8192, 256, 0, stream>>>(x, n1w, n1b, hbuf);
  gemm_qkv<<<576, 256, 0, stream>>>(hbuf, wqkvb, qkvb);   // 128*36 wave tiles / 8
  qm_gemm<<<192, 256, 0, stream>>>(qkvb, MTb, qMb, 0);    // qM: 32*48 / 8
  qm_gemm<<<192, 256, 0, stream>>>(qkvb, MTb, kMb, 1);    // kM
  kmk_dot<<<4096, 256, 0, stream>>>(qkvb, kMb, kMkb);
  v_transpose<<<32 * 64 * 12, 256, 0, stream>>>(qkvb, vTb);
  flash_attn<<<256, 256, 0, stream>>>(qkvb, qMb, kMkb, vTb, scale, ybuf);
  gemm_proj<<<192, 256, 0, stream>>>(ybuf, wprjb, proj_b, x, out);
  ln2_rbf<<<8192, 256, 0, stream>>>(out, n2w, n2b, centers, beta, fc_w);
}
